// RelGraphSAGE_1958505087380
// MI455X (gfx1250) — matrix operand order repaired and checked by reading, not compiled, not run on hardware
//
#include <hip/hip_runtime.h>
#include <hip/hip_bf16.h>

typedef __bf16 bf16;
typedef __attribute__((ext_vector_type(16))) __bf16 v16bf;
typedef __attribute__((ext_vector_type(8)))  __bf16 v8bf;
typedef __attribute__((ext_vector_type(8)))  float   v8f;

#define HIDC 64
#define HCC  256
#define NHEAD 4
#define EDIMC 6

// ------------------------- utility kernels -------------------------
__global__ void k_fill_f32(float* p, float v, int n) {
  int i = blockIdx.x * blockDim.x + threadIdx.x;
  if (i < n) p[i] = v;
}
__global__ void k_fill_u32(unsigned* p, unsigned v, int n) {
  int i = blockIdx.x * blockDim.x + threadIdx.x;
  if (i < n) p[i] = v;
}
__global__ void k_copy_f32(const float* __restrict__ a, float* __restrict__ b, int n) {
  int i = blockIdx.x * blockDim.x + threadIdx.x;
  if (i < n) b[i] = a[i];
}
__global__ void k_relu(const float* __restrict__ a, float* __restrict__ b, int n) {
  int i = blockIdx.x * blockDim.x + threadIdx.x;
  if (i < n) b[i] = fmaxf(a[i], 0.0f);
}
__global__ void k_f2bf(const float* __restrict__ a, bf16* __restrict__ b, int n) {
  int i = blockIdx.x * blockDim.x + threadIdx.x;
  if (i < n) b[i] = (bf16)a[i];
}

// initial action-node features: x@W + b + embT[tix] + embB[bix]
__global__ void k_init_action(const float* __restrict__ x, const int* __restrict__ tix,
                              const int* __restrict__ bix, const float* __restrict__ W,
                              const float* __restrict__ b, const float* __restrict__ embT,
                              const float* __restrict__ embB, float* __restrict__ h, int N) {
  int i = blockIdx.x * blockDim.x + threadIdx.x;
  if (i >= N * HIDC) return;
  int n = i >> 6, c = i & 63;
  float acc = b[c];
#pragma unroll
  for (int d = 0; d < 5; ++d) acc += x[(size_t)n * 5 + d] * W[d * HIDC + c];
  acc += embT[(size_t)tix[n] * HIDC + c] + embB[(size_t)bix[n] * HIDC + c];
  h[i] = acc;
}

// Pack fp32 weight [Kdim,Ncols] into WMMA B-fragment layout (bf16):
// P[((t*nkb+kb)*32 + lane)*16 + j] = W[(kb*32 + (j&7) + (j>=8?16:0) + (lane>=16?8:0)) * Ncols + t*16 + (lane&15)]
__global__ void k_pack_w(const float* __restrict__ W, bf16* __restrict__ P, int Kdim, int Ncols) {
  int i = blockIdx.x * blockDim.x + threadIdx.x;
  if (i >= Kdim * Ncols) return;
  int j = i & 15;
  int lane = (i >> 4) & 31;
  int rest = i >> 9;
  int nkb = Kdim >> 5;
  int t = rest / nkb, kb = rest - t * nkb;
  int n = t * 16 + (lane & 15);
  int K = kb * 32 + (j & 7) + ((j >> 3) << 4) + ((lane >> 4) << 3);
  P[i] = (bf16)W[(size_t)K * Ncols + n];
}

// ------------------------- WMMA GEMM -------------------------
// Y[N,Ncols] (+)= Xbf[N,64] @ Wpacked + bias.
// One wave handles FOUR 16-row tiles across all 16-wide column tiles:
// each B-fragment pair is reused by 4 WMMAs (8 v_wmma per B load pair).
// All row-group/tile-validity checks are wave-uniform -> EXEC all-ones at WMMA.
__global__ __launch_bounds__(256)
void k_gemm_bf16_wmma(const bf16* __restrict__ X, const bf16* __restrict__ Wp,
                      const float* __restrict__ bias, float* __restrict__ Y,
                      int N, int Ncols, int accum) {
  const int lane = threadIdx.x & 31;
  const int halfHi = lane >> 4;               // 0: K-low half, 1: K-high half
  const int rowTiles = (N + 15) >> 4;
  const int rt0 = (blockIdx.x * 8 + (threadIdx.x >> 5)) * 4;
  if (rt0 >= rowTiles) return;                // wave-uniform

  // A fragments per CDNA5 16-bit 16x32 layout:
  // lanes 0-15: K = 0..7,16..23 ; lanes 16-31: K = 8..15,24..31 (+32 for kb=1)
  v16bf a[4][2];
#pragma unroll
  for (int rg = 0; rg < 4; ++rg) {
    int rt = rt0 + rg; if (rt >= rowTiles) rt = rowTiles - 1;   // safe dup loads
    int arow = rt * 16 + (lane & 15); if (arow >= N) arow = N - 1;
    const bf16* Xr = X + (size_t)arow * HIDC;
#pragma unroll
    for (int kb = 0; kb < 2; ++kb) {
      int base = kb * 32 + halfHi * 8;
      v8bf lo = *(const v8bf*)(Xr + base);
      v8bf hi = *(const v8bf*)(Xr + base + 16);
      a[rg][kb] = __builtin_shufflevector(lo, hi, 0,1,2,3,4,5,6,7,8,9,10,11,12,13,14,15);
    }
  }

  const int nT = Ncols >> 4;
  for (int t = 0; t < nT; ++t) {
    v16bf b0 = *(const v16bf*)(Wp + (((size_t)(t * 2 + 0) * 32 + lane) << 4));
    v16bf b1 = *(const v16bf*)(Wp + (((size_t)(t * 2 + 1) * 32 + lane) << 4));
    const int col = t * 16 + (lane & 15);
    const float bv = bias[col];
#pragma unroll
    for (int rg = 0; rg < 4; ++rg) {
      const int rt = rt0 + rg;
      if (rt >= rowTiles) break;              // wave-uniform
      v8f c = {};
      c = __builtin_amdgcn_wmma_f32_16x16x32_bf16(false, a[rg][0], false, b0, (short)0, c, false, false);
      c = __builtin_amdgcn_wmma_f32_16x16x32_bf16(false, a[rg][1], false, b1, (short)0, c, false, false);
      // C/D layout: lane -> N=lane&15 ; VGPR i -> M = i (+8 for hi lanes)
      const int mbase = rt * 16 + halfHi * 8;
      float* Yp = Y + (size_t)mbase * Ncols + col;
      const bool fullTile = (rt * 16 + 16) <= N;   // wave-uniform
      if (fullTile) {
        if (accum) {
          // batch the 8 reads first so they issue as one clause (no serialized RMW chain)
          float oldv[8];
#pragma unroll
          for (int i = 0; i < 8; ++i) oldv[i] = Yp[(size_t)i * Ncols];
#pragma unroll
          for (int i = 0; i < 8; ++i) Yp[(size_t)i * Ncols] = oldv[i] + c[i] + bv;
        } else {
#pragma unroll
          for (int i = 0; i < 8; ++i) Yp[(size_t)i * Ncols] = c[i] + bv;
        }
      } else {
#pragma unroll
        for (int i = 0; i < 8; ++i) {
          if (mbase + i < N) {
            float val = c[i] + bv;
            if (accum) Yp[(size_t)i * Ncols] += val; else Yp[(size_t)i * Ncols] = val;
          }
        }
      }
    }
  }
}

// ------------------------- edge kernels -------------------------
__device__ __forceinline__ unsigned f32_ord(float f) {
  unsigned u = __float_as_uint(f);
  return (u & 0x80000000u) ? ~u : (u | 0x80000000u);
}
__device__ __forceinline__ float ord_f32(unsigned u) {
  return (u & 0x80000000u) ? __uint_as_float(u & 0x7FFFFFFFu) : __uint_as_float(~u);
}

// pass 1: logits[e,h] = dot(q[dst], k[src]+e_proj)/8 ; atomic max per (dst,h)
__global__ __launch_bounds__(256)
void k_edge_logits(const float* __restrict__ q, const float* __restrict__ kmat,
                   const float* __restrict__ ea, const float* __restrict__ We,
                   const float* __restrict__ be, const int* __restrict__ src,
                   const int* __restrict__ dst, float* __restrict__ logits,
                   unsigned* __restrict__ mx, int E) {
  __shared__ float sWe[EDIMC * HCC];
  __shared__ float sbe[HCC];
  for (int i = threadIdx.x; i < EDIMC * HCC; i += blockDim.x) sWe[i] = We[i];
  for (int i = threadIdx.x; i < HCC; i += blockDim.x) sbe[i] = be[i];
  __syncthreads();
  int e = (blockIdx.x * blockDim.x + threadIdx.x) >> 5;
  int lane = threadIdx.x & 31;
  if (e >= E) return;
  int s = src[e], d = dst[e];
  float ew[EDIMC];
#pragma unroll
  for (int j = 0; j < EDIMC; ++j) ew[j] = ea[(size_t)e * EDIMC + j];
  const float* qr = q + (size_t)d * HCC;
  const float* kr = kmat + (size_t)s * HCC;
  float part = 0.0f;
#pragma unroll
  for (int u = 0; u < 8; ++u) {
    int c = lane * 8 + u;
    float ev = sbe[c];
#pragma unroll
    for (int j = 0; j < EDIMC; ++j) ev += ew[j] * sWe[j * HCC + c];
    part += qr[c] * (kr[c] + ev);
  }
  part += __shfl_xor(part, 1, 32);
  part += __shfl_xor(part, 2, 32);
  part += __shfl_xor(part, 4, 32);
  if ((lane & 7) == 0) {
    int h = lane >> 3;
    float lg = part * 0.125f;  // 1/sqrt(64)
    logits[(size_t)e * NHEAD + h] = lg;
    atomicMax(&mx[(size_t)d * NHEAD + h], f32_ord(lg));
  }
}

// pass 2: ex = exp(logit - m[dst]); den[dst,h] += ex (in-place into logits buf)
__global__ void k_edge_den(const int* __restrict__ dst, float* __restrict__ logits,
                           const unsigned* __restrict__ mx, float* __restrict__ den, int E) {
  int i = blockIdx.x * blockDim.x + threadIdx.x;
  if (i >= E * NHEAD) return;
  int e = i >> 2, h = i & 3;
  int d = dst[e];
  float m = ord_f32(mx[(size_t)d * NHEAD + h]);
  float ex = __expf(logits[i] - m);
  logits[i] = ex;
  atomicAdd(&den[(size_t)d * NHEAD + h], ex);
}

// pass 3: agg[dst,c] += sum_h (ex/den) * 0.25 * (v[src,h*64+c] + e_proj)
__global__ __launch_bounds__(256)
void k_edge_scatter(const float* __restrict__ vmat, const float* __restrict__ ea,
                    const float* __restrict__ We, const float* __restrict__ be,
                    const int* __restrict__ src, const int* __restrict__ dst,
                    const float* __restrict__ exbuf, const float* __restrict__ den,
                    float* __restrict__ agg, int E) {
  __shared__ float sWe[EDIMC * HCC];
  __shared__ float sbe[HCC];
  for (int i = threadIdx.x; i < EDIMC * HCC; i += blockDim.x) sWe[i] = We[i];
  for (int i = threadIdx.x; i < HCC; i += blockDim.x) sbe[i] = be[i];
  __syncthreads();
  int e = (blockIdx.x * blockDim.x + threadIdx.x) >> 5;
  int lane = threadIdx.x & 31;
  if (e >= E) return;
  int s = src[e], d = dst[e];
  float ew[EDIMC];
#pragma unroll
  for (int j = 0; j < EDIMC; ++j) ew[j] = ea[(size_t)e * EDIMC + j];
  float alpha[NHEAD];
#pragma unroll
  for (int h = 0; h < NHEAD; ++h)
    alpha[h] = exbuf[(size_t)e * NHEAD + h] / den[(size_t)d * NHEAD + h] * 0.25f;
  const float* vr = vmat + (size_t)s * HCC;
#pragma unroll
  for (int u = 0; u < 2; ++u) {
    int c = lane * 2 + u;
    float acc = 0.0f;
#pragma unroll
    for (int h = 0; h < NHEAD; ++h) {
      int ch = h * HIDC + c;
      float ev = sbe[ch];
#pragma unroll
      for (int j = 0; j < EDIMC; ++j) ev += ew[j] * sWe[j * HCC + ch];
      acc += alpha[h] * (vr[ch] + ev);
    }
    atomicAdd(&agg[(size_t)d * HIDC + c], acc);
  }
}

// ------------------------- host orchestration -------------------------
static inline int cdiv(long long a, long long b) { return (int)((a + b - 1) / b); }

extern "C" void kernel_launch(void* const* d_in, const int* in_sizes, int n_in,
                              void* d_out, int out_size, void* d_ws, size_t ws_size,
                              hipStream_t stream) {
  (void)in_sizes; (void)n_in; (void)out_size; (void)ws_size;
  const int NS = 200000, NE_ = 200000, NP = 20000, NT = 1000;
  const int Eb = 200000, Etp = 20000, Lnum = 2;

  const float* x_start = (const float*)d_in[0];
  const float* x_end   = (const float*)d_in[1];
  const int* s_tix = (const int*)d_in[2];
  const int* s_bix = (const int*)d_in[3];
  const int* e_tix = (const int*)d_in[4];
  const int* e_bix = (const int*)d_in[5];
  const int* src_arr[4] = { (const int*)d_in[6], (const int*)d_in[8],
                            (const int*)d_in[10], (const int*)d_in[12] };
  const int* dst_arr[4] = { (const int*)d_in[7], (const int*)d_in[9],
                            (const int*)d_in[11], (const int*)d_in[13] };
  const float* ea_arr[4] = { (const float*)d_in[14], (const float*)d_in[15],
                             (const float*)d_in[16], (const float*)d_in[17] };
  const float* emb_st = (const float*)d_in[18];
  const float* emb_sb = (const float*)d_in[19];
  const float* emb_et = (const float*)d_in[20];
  const float* emb_eb = (const float*)d_in[21];
  const float* emb_pl = (const float*)d_in[22];
  const float* emb_tm = (const float*)d_in[23];
  const float* W_start = (const float*)d_in[24];
  const float* b_start = (const float*)d_in[25];
  const float* W_end   = (const float*)d_in[26];
  const float* b_end   = (const float*)d_in[27];
  const float* Wq = (const float*)d_in[28];  const float* bq = (const float*)d_in[29];
  const float* Wk = (const float*)d_in[30];  const float* bk = (const float*)d_in[31];
  const float* Wv = (const float*)d_in[32];  const float* bv = (const float*)d_in[33];
  const float* We = (const float*)d_in[34];  const float* be = (const float*)d_in[35];
  const float* Wsk = (const float*)d_in[36]; const float* bsk = (const float*)d_in[37];
  const float* W_head = (const float*)d_in[38];
  const float* b_head = (const float*)d_in[39];

  const int rel_s[4]  = { 2, 0, 1, 3 };
  const int rel_d[4]  = { 0, 1, 0, 2 };
  const int rel_E[4]  = { Eb, Eb, Eb, Etp };
  const int ncnt[4]   = { NS, NE_, NP, NT };

  char* ws = (char*)d_ws;
  size_t off = 0;
  auto alloc = [&](size_t bytes) -> char* {
    char* p = ws + off;
    off = (off + bytes + 255) & ~(size_t)255;
    return p;
  };
  float* h[4];   for (int i = 0; i < 4; ++i) h[i]  = (float*)alloc((size_t)ncnt[i] * 64 * 4);
  float* aggb[3];for (int i = 0; i < 3; ++i) aggb[i] = (float*)alloc((size_t)ncnt[i] * 64 * 4);
  bf16* hb[4];   for (int i = 0; i < 4; ++i) hb[i] = (bf16*)alloc((size_t)ncnt[i] * 64 * 2);
  float* qbuf = (float*)alloc((size_t)200000 * 256 * 4);
  float* kbuf = (float*)alloc((size_t)200000 * 256 * 4);
  float* vbuf = (float*)alloc((size_t)200000 * 256 * 4);
  float* lg   = (float*)alloc((size_t)200000 * 4 * 4);
  unsigned* mx = (unsigned*)alloc((size_t)200000 * 4 * 4);
  float* den  = (float*)alloc((size_t)200000 * 4 * 4);
  bf16* wq_p = (bf16*)alloc((size_t)Lnum * 4 * 64 * 256 * 2);
  bf16* wk_p = (bf16*)alloc((size_t)Lnum * 4 * 64 * 256 * 2);
  bf16* wv_p = (bf16*)alloc((size_t)Lnum * 4 * 64 * 256 * 2);
  bf16* ws_p = (bf16*)alloc((size_t)Lnum * 4 * 64 * 64 * 2);
  bf16* wh_p = (bf16*)alloc((size_t)64 * 32 * 2);

  const int TB = 256;
  // -------- pack all weights into bf16 WMMA fragments --------
  for (int s = 0; s < Lnum * 4; ++s) {
    k_pack_w<<<cdiv(64 * 256, TB), TB, 0, stream>>>(Wq + (size_t)s * 64 * 256, wq_p + (size_t)s * 64 * 256, 64, 256);
    k_pack_w<<<cdiv(64 * 256, TB), TB, 0, stream>>>(Wk + (size_t)s * 64 * 256, wk_p + (size_t)s * 64 * 256, 64, 256);
    k_pack_w<<<cdiv(64 * 256, TB), TB, 0, stream>>>(Wv + (size_t)s * 64 * 256, wv_p + (size_t)s * 64 * 256, 64, 256);
    k_pack_w<<<cdiv(64 * 64, TB), TB, 0, stream>>>(Wsk + (size_t)s * 64 * 64, ws_p + (size_t)s * 64 * 64, 64, 64);
  }
  k_pack_w<<<cdiv(64 * 32, TB), TB, 0, stream>>>(W_head, wh_p, 64, 32);

  // -------- initial node features --------
  k_init_action<<<cdiv((long long)NS * 64, TB), TB, 0, stream>>>(x_start, s_tix, s_bix, W_start, b_start, emb_st, emb_sb, h[0], NS);
  k_init_action<<<cdiv((long long)NE_ * 64, TB), TB, 0, stream>>>(x_end, e_tix, e_bix, W_end, b_end, emb_et, emb_eb, h[1], NE_);
  k_copy_f32<<<cdiv((long long)NP * 64, TB), TB, 0, stream>>>(emb_pl, h[2], NP * 64);
  k_copy_f32<<<cdiv((long long)NT * 64, TB), TB, 0, stream>>>(emb_tm, h[3], NT * 64);

  auto gemm_blocks = [&](int N) { return cdiv(cdiv(N, 16), 8 * 4); };

  // -------- layers --------
  for (int l = 0; l < Lnum; ++l) {
    for (int i = 0; i < 4; ++i)
      k_f2bf<<<cdiv((long long)ncnt[i] * 64, TB), TB, 0, stream>>>(h[i], hb[i], ncnt[i] * 64);
    for (int i = 0; i < 3; ++i)
      k_fill_f32<<<cdiv((long long)ncnt[i] * 64, TB), TB, 0, stream>>>(aggb[i], 0.0f, ncnt[i] * 64);

    for (int r = 0; r < 4; ++r) {
      const int sl = l * 4 + r;
      const int snt = rel_s[r], dnt = rel_d[r];
      const int Ns = ncnt[snt], Nd = ncnt[dnt], E = rel_E[r];
      k_gemm_bf16_wmma<<<gemm_blocks(Nd), TB, 0, stream>>>(hb[dnt], wq_p + (size_t)sl * 16384, bq + (size_t)sl * 256, qbuf, Nd, 256, 0);
      k_gemm_bf16_wmma<<<gemm_blocks(Ns), TB, 0, stream>>>(hb[snt], wk_p + (size_t)sl * 16384, bk + (size_t)sl * 256, kbuf, Ns, 256, 0);
      k_gemm_bf16_wmma<<<gemm_blocks(Ns), TB, 0, stream>>>(hb[snt], wv_p + (size_t)sl * 16384, bv + (size_t)sl * 256, vbuf, Ns, 256, 0);
      k_fill_u32<<<cdiv((long long)Nd * 4, TB), TB, 0, stream>>>(mx, 0x007FFFFFu /* ord(-inf) */, Nd * 4);
      k_fill_f32<<<cdiv((long long)Nd * 4, TB), TB, 0, stream>>>(den, 0.0f, Nd * 4);
      k_edge_logits<<<cdiv(E, 8), TB, 0, stream>>>(qbuf, kbuf, ea_arr[r], We + (size_t)sl * 1536, be + (size_t)sl * 256,
                                                   src_arr[r], dst_arr[r], lg, mx, E);
      k_edge_den<<<cdiv((long long)E * 4, TB), TB, 0, stream>>>(dst_arr[r], lg, mx, den, E);
      k_edge_scatter<<<cdiv(E, 8), TB, 0, stream>>>(vbuf, ea_arr[r], We + (size_t)sl * 1536, be + (size_t)sl * 256,
                                                    src_arr[r], dst_arr[r], lg, den, aggb[dnt], E);
      k_gemm_bf16_wmma<<<gemm_blocks(Nd), TB, 0, stream>>>(hb[dnt], ws_p + (size_t)sl * 4096, bsk + (size_t)sl * 64, aggb[dnt], Nd, 64, 1);
    }
    for (int i = 0; i < 3; ++i)
      k_relu<<<cdiv((long long)ncnt[i] * 64, TB), TB, 0, stream>>>(aggb[i], h[i], ncnt[i] * 64);
    k_relu<<<cdiv((long long)NT * 64, TB), TB, 0, stream>>>(h[3], h[3], NT * 64);
  }

  // -------- output head: [NE,64] @ [64,32] + b --------
  k_f2bf<<<cdiv((long long)NE_ * 64, TB), TB, 0, stream>>>(h[1], hb[1], NE_ * 64);
  k_gemm_bf16_wmma<<<gemm_blocks(NE_), TB, 0, stream>>>(hb[1], wh_p, b_head, (float*)d_out, NE_, 32, 0);
}
